// AttnLSTMDecoder_8555574854484
// MI455X (gfx1250) — matrix operand order repaired and checked by reading, not compiled, not run on hardware
//
#include <hip/hip_runtime.h>
#include <hip/hip_bf16.h>

// ---------------------------------------------------------------------------
// AttnLSTM decoder for MI455X (gfx1250, wave32, WMMA + TDM async tensor DMA).
// All GEMMs use v_wmma_f32_16x16x32_bf16 (bf16 in, f32 accumulate).
// A-panels are staged into LDS with tensor_load_to_lds where layout permits.
// ---------------------------------------------------------------------------

typedef __attribute__((ext_vector_type(16))) __bf16 v16bf;
typedef __attribute__((ext_vector_type(8)))  __bf16 v8bf;
typedef __attribute__((ext_vector_type(8)))  float  v8f;

static constexpr int B_   = 32;
static constexpr int T_   = 64;
static constexpr int S_   = 64;
static constexpr int H_   = 512;
static constexpr int E_   = 256;
static constexpr int E2_  = 2 * H_;    // 1024
static constexpr int G4_  = 4 * H_;    // 2048
static constexpr int VOUT_ = 32000;
static constexpr int BT_  = B_ * T_;   // 2048
static constexpr int BS_  = B_ * S_;   // 2048
static constexpr int HC_  = 3 * H_;    // 1536

// ----- Tensor Data Mover: DMA a [rows x kelems] bf16 row-major panel -------
#if defined(__AMDGCN__) && __has_builtin(__builtin_amdgcn_tensor_load_to_lds)
#define HAVE_TDM 1
typedef unsigned v4u __attribute__((ext_vector_type(4)));
typedef int      v8i __attribute__((ext_vector_type(8)));
typedef int      v4i __attribute__((ext_vector_type(4)));

__device__ __forceinline__ unsigned lds_off(const void* p) {
  // generic -> LDS addrspace cast strips the aperture; value = byte offset
  return (unsigned)(unsigned long long)(const __attribute__((address_space(3))) void*)p;
}

// D# per CDNA5 ISA 8.3/8.4: 2-D tile, contiguous rows (stride == kelems).
// This toolchain exposes the 6-arg builtin: (g0 v4u, g1 v8i, g2 v4i, g3 v4i,
// extra v8i, cpol). Groups 2/3 + trailing words are zero for a 2-D tile.
__device__ __forceinline__ void tdm_load_panel_bf16(const __bf16* gptr,
                                                    unsigned lds_byte,
                                                    int kelems, int rows) {
  unsigned long long ga = (unsigned long long)(uintptr_t)gptr;
  v4u g0;
  g0[0] = 1u;                                  // count=1, user descriptor
  g0[1] = lds_byte;                            // lds_addr (bytes)
  g0[2] = (unsigned)ga;                        // global_addr[31:0]
  g0[3] = (unsigned)((ga >> 32) & 0x01ffffffu) // global_addr[56:32]
        | (2u << 30);                          // type=2 ("image")
  v8i g1;
  g1[0] = (int)(1u << 16);                     // data_size=1 -> 2 bytes
  g1[1] = (int)((unsigned)kelems << 16);       // tensor_dim0[15:0]
  g1[2] = (int)((((unsigned)kelems >> 16) & 0xffffu)
        | ((unsigned)rows << 16));             // dim0[31:16] | tensor_dim1[15:0]
  g1[3] = (int)((unsigned)kelems << 16);       // dim1[31:16]=0 | tile_dim0
  g1[4] = rows;                                // tile_dim1 | tile_dim2=0
  g1[5] = kelems;                              // tensor_dim0_stride[31:0]
  g1[6] = 0;                                   // stride[47:32] | dim1_stride lo
  g1[7] = 0;
  v4i z4 = {0, 0, 0, 0};
  v8i z8 = {0, 0, 0, 0, 0, 0, 0, 0};
  __builtin_amdgcn_tensor_load_to_lds(g0, g1, z4, z4, z8, 0);
}
#endif

// ----- WMMA fragment loaders (16-bit A 16x32 / B 32x16 ISA layouts) --------
__device__ __forceinline__ v16bf load_a_frag(const __bf16* A, int lda,
                                             int m0, int k0, int lane) {
  int row = m0 + (lane & 15);
  int kk  = k0 + ((lane >> 4) << 3);
  const __bf16* p = A + (long long)row * lda + kk;
  v8bf lo = *(const v8bf*)(p);
  v8bf hi = *(const v8bf*)(p + 16);
  return __builtin_shufflevector(lo, hi, 0,1,2,3,4,5,6,7,8,9,10,11,12,13,14,15);
}

__device__ __forceinline__ v16bf load_b_frag(const __bf16* W, int ldw,
                                             int n0, int k0, int lane) {
  int row = n0 + (lane & 15);
  int kk  = k0 + ((lane >> 4) << 3);
  const __bf16* p = W + (long long)row * ldw + kk;
  v8bf lo = *(const v8bf*)(p);
  v8bf hi = *(const v8bf*)(p + 16);
  return __builtin_shufflevector(lo, hi, 0,1,2,3,4,5,6,7,8,9,10,11,12,13,14,15);
}

__device__ __forceinline__ v8f wmma_bf16(v16bf a, v16bf b, v8f c) {
  return __builtin_amdgcn_wmma_f32_16x16x32_bf16(
      false, a, false, b, (short)0, c, false, false);
}

__device__ __forceinline__ float sigmoidf_(float x) {
  return 1.0f / (1.0f + __expf(-x));
}

// One K-loop instantiation. Inlined per call site so the A pointer keeps its
// address space (LDS vs global) and each loop stays branch-free inside.
__device__ __forceinline__ void gemm_kloop(const __bf16* Aptr, int alda, int am0,
                                           const __bf16* Wb, int ldw, int nt,
                                           int K, int lane, v8f acc[4]) {
  for (int k0 = 0; k0 < K; k0 += 32) {
    if (k0 + 32 < K) {
      __builtin_prefetch(Wb + (long long)(nt * 16 + (lane & 15)) * ldw + k0 + 32, 0, 1);
    }
    v16bf bf = load_b_frag(Wb, ldw, nt * 16, k0, lane);
#pragma unroll
    for (int i = 0; i < 4; ++i) {
      v16bf af = load_a_frag(Aptr, alda, am0 + i * 16, k0, lane);
      acc[i] = wmma_bf16(af, bf, acc[i]);
    }
  }
}

// ----- generic batched GEMM: C[b] = act(A[b] (MxK) . W[b] (NxK)^T) ---------
// Block = 256 threads = 8 waves. Wave computes a 64x16 strip (4 M-tiles share
// each B fragment). All 8 waves share one 64-row A panel; if A is dense
// (lda==K, K<=512) the panel is staged into LDS by the Tensor Data Mover.
// grid = (ceil(N/128), M/64, batch); requires M%64==0, N%16==0, K%32==0.
__global__ void __launch_bounds__(256)
wmma_gemm_bf16(const __bf16* __restrict__ A,
               const __bf16* __restrict__ W,
               float* __restrict__ Cf,   // may be null
               __bf16* __restrict__ Cb,  // may be null
               int M, int N, int K, int lda, int ldw,
               int ldcf, int ldcb,
               long long sA, long long sW,
               long long sCf, long long sCb,
               int act) {
  __shared__ __bf16 As[64 * 512];    // 64 KB A panel
  int lane = threadIdx.x & 31;
  int wv   = threadIdx.x >> 5;       // 0..7
  int ntn  = N >> 4;
  int nt   = blockIdx.x * 8 + wv;
  int m0   = blockIdx.y * 64;
  int b    = blockIdx.z;
  const __bf16* Ab = A + (long long)b * sA;
  const __bf16* Wb = W + (long long)b * sW;

  bool use_lds = false;
#if defined(HAVE_TDM)
  use_lds = (lda == K) && (K <= 512);
  if (use_lds) {
    if (wv == 0) {
      tdm_load_panel_bf16(Ab + (long long)m0 * lda, lds_off(As), K, 64);
      __builtin_amdgcn_s_wait_tensorcnt(0);
    }
    __syncthreads();
  }
#endif
  if (nt >= ntn) return;

  v8f acc[4] = {{}, {}, {}, {}};
  // Uniform branch hoisted OUT of the K loop: two clean loop bodies, one with
  // ds_load_b128 A-fragments (TDM-staged panel), one with global A loads.
  if (use_lds) {
    gemm_kloop(As, K, 0, Wb, ldw, nt, K, lane, acc);
  } else {
    gemm_kloop(Ab, lda, m0, Wb, ldw, nt, K, lane, acc);
  }

  int n = nt * 16 + (lane & 15);
#pragma unroll
  for (int i = 0; i < 4; ++i) {
    int m_base = m0 + i * 16 + ((lane >> 4) << 3);
#pragma unroll
    for (int r = 0; r < 8; ++r) {
      float v = acc[i][r];
      if (act == 1) v = tanhf(v);
      if (Cf) Cf[(long long)b * sCf + (long long)(m_base + r) * ldcf + n] = v;
      if (Cb) Cb[(long long)b * sCb + (long long)(m_base + r) * ldcb + n] = (__bf16)v;
    }
  }
}

// ----- fused LSTM step: g = Xp[t] + h_in.Whh^T + b; gates; c,h update ------
// grid = (H/16, B/16), block = 32 (one wave); wave owns 16 batches x 16 hidden.
// The 16x512 h panel is DMAed to LDS by the TDM (recurrence is latency-bound;
// one bulk async transfer replaces scattered per-lane loads).
__global__ void __launch_bounds__(32)
lstm_step(const __bf16* __restrict__ h_in,   // [B,H] bf16
          __bf16* __restrict__ h_out,        // [B,H] bf16
          float* __restrict__ hf32,          // [B,H] f32
          float* __restrict__ c,             // [B,H] f32
          __bf16* __restrict__ h_all,        // [B*T,H] bf16
          const float* __restrict__ Xp,      // [B*T,4H] f32
          const __bf16* __restrict__ Whh,    // [4H,H] bf16
          const float* __restrict__ b_ih,
          const float* __restrict__ b_hh,
          int t) {
  __shared__ __bf16 Hs[16 * H_];     // 16 KB h panel
  int lane = threadIdx.x & 31;
  int nt = blockIdx.x;   // hidden tile
  int mt = blockIdx.y;   // batch tile

  const __bf16* Ap = h_in;
  int am0 = mt * 16;
#if defined(HAVE_TDM)
  tdm_load_panel_bf16(h_in + (long long)mt * 16 * H_, lds_off(Hs), H_, 16);
  __builtin_amdgcn_s_wait_tensorcnt(0);
  Ap = Hs;
  am0 = 0;
#endif

  v8f acc[4] = {{}, {}, {}, {}};
  for (int k0 = 0; k0 < H_; k0 += 32) {
    v16bf af = load_a_frag(Ap, H_, am0, k0, lane);
#pragma unroll
    for (int q = 0; q < 4; ++q) {
      v16bf bf = load_b_frag(Whh, H_, q * H_ + nt * 16, k0, lane);
      acc[q] = wmma_bf16(af, bf, acc[q]);
    }
  }

  int m_base = mt * 16 + ((lane >> 4) << 3);
  int n = nt * 16 + (lane & 15);
#pragma unroll
  for (int r = 0; r < 8; ++r) {
    int bb = m_base + r;
    long long xrow = (long long)(bb * T_ + t) * G4_;
    float gi = acc[0][r] + Xp[xrow + 0 * H_ + n] + b_ih[0 * H_ + n] + b_hh[0 * H_ + n];
    float gf = acc[1][r] + Xp[xrow + 1 * H_ + n] + b_ih[1 * H_ + n] + b_hh[1 * H_ + n];
    float gg = acc[2][r] + Xp[xrow + 2 * H_ + n] + b_ih[2 * H_ + n] + b_hh[2 * H_ + n];
    float go = acc[3][r] + Xp[xrow + 3 * H_ + n] + b_ih[3 * H_ + n] + b_hh[3 * H_ + n];
    float ig = sigmoidf_(gi), fg = sigmoidf_(gf);
    float gc = tanhf(gg),     og = sigmoidf_(go);
    long long ci = (long long)bb * H_ + n;
    float cn = fg * c[ci] + ig * gc;
    float hn = og * tanhf(cn);
    c[ci] = cn;
    hf32[ci] = hn;
    h_out[ci] = (__bf16)hn;
    h_all[(long long)(bb * T_ + t) * H_ + n] = (__bf16)hn;
  }
}

// ----- small helpers -------------------------------------------------------
__global__ void cast_f32_bf16(const float* __restrict__ src,
                              __bf16* __restrict__ dst, long long n) {
  long long i = blockIdx.x * 256LL + threadIdx.x;
  if (i < n) dst[i] = (__bf16)src[i];
}

__global__ void embed_gather(const int* __restrict__ ids,
                             const float* __restrict__ emb,
                             __bf16* __restrict__ x) {
  int row = blockIdx.x;            // b*T + t
  int e = threadIdx.x;             // 0..E-1
  long long id = ids[row];
  x[(long long)row * E_ + e] = (__bf16)emb[id * E_ + e];
}

__global__ void init_state(const float* __restrict__ h0,
                           __bf16* __restrict__ hping,
                           float* __restrict__ hf32,
                           float* __restrict__ c) {
  int i = blockIdx.x * 256 + threadIdx.x;
  if (i < B_ * H_) {
    float v = h0[i];
    hping[i] = (__bf16)v;
    hf32[i] = v;
    c[i] = 0.0f;
  }
}

__global__ void softmax_rows64(const float* __restrict__ scores,
                               __bf16* __restrict__ attn) {
  int row = blockIdx.x;            // B*T rows
  int tid = threadIdx.x;           // 64
  __shared__ float sm[64];
  float v = scores[(long long)row * S_ + tid];
  sm[tid] = v; __syncthreads();
  for (int s = 32; s > 0; s >>= 1) {
    if (tid < s) sm[tid] = fmaxf(sm[tid], sm[tid + s]);
    __syncthreads();
  }
  float m = sm[0]; __syncthreads();
  float e = __expf(v - m);
  sm[tid] = e; __syncthreads();
  for (int s = 32; s > 0; s >>= 1) {
    if (tid < s) sm[tid] += sm[tid + s];
    __syncthreads();
  }
  attn[(long long)row * S_ + tid] = (__bf16)(e / sm[0]);
}

__global__ void transpose_enc(const __bf16* __restrict__ enc,  // [B,S,E2]
                              __bf16* __restrict__ encT) {     // [B,E2,S]
  long long idx = blockIdx.x * 256LL + threadIdx.x;
  if (idx >= (long long)B_ * S_ * E2_) return;
  int e = (int)(idx & (E2_ - 1));
  long long r = idx >> 10;
  int s = (int)(r & (S_ - 1));
  int b = (int)(r >> 6);
  encT[((long long)b * E2_ + e) * S_ + s] = enc[idx];
}

__global__ void copy_out_to_hc(const __bf16* __restrict__ out,  // [B*T,H]
                               __bf16* __restrict__ hc) {       // [B*T,3H]
  long long idx = blockIdx.x * 256LL + threadIdx.x;
  if (idx >= (long long)BT_ * H_) return;
  int col = (int)(idx & (H_ - 1));
  long long row = idx >> 9;
  hc[row * HC_ + col] = out[idx];
}

__global__ void log_softmax_rows(float* __restrict__ x, int N) {
  int row = blockIdx.x;
  float* p = x + (long long)row * N;
  int tid = threadIdx.x;           // 256
  float m = -3.0e38f, s = 0.0f;
  for (int i = tid; i < N; i += 256) {
    float v = p[i];
    if (v > m) { s = s * __expf(m - v) + 1.0f; m = v; }
    else       { s += __expf(v - m); }
  }
  __shared__ float sm[256], ss[256];
  sm[tid] = m; ss[tid] = s; __syncthreads();
  if (tid == 0) {
    float M = sm[0], Ssum = ss[0];
    for (int i = 1; i < 256; ++i) {
      float mi = sm[i], si = ss[i];
      if (mi > M) { Ssum = Ssum * __expf(M - mi) + si; M = mi; }
      else        { Ssum += si * __expf(mi - M); }
    }
    sm[0] = M; ss[0] = Ssum;
  }
  __syncthreads();
  float lse = sm[0] + __logf(ss[0]);
  for (int i = tid; i < N; i += 256) p[i] -= lse;
}

__global__ void write_hout(const float* __restrict__ h0,
                           const float* __restrict__ h1,
                           float* __restrict__ dst) {
  int i = blockIdx.x * 256 + threadIdx.x;
  if (i < B_ * H_) {
    dst[i] = h0[i];
    dst[B_ * H_ + i] = h1[i];
  }
}

// ---------------------------------------------------------------------------
extern "C" void kernel_launch(void* const* d_in, const int* in_sizes, int n_in,
                              void* d_out, int out_size, void* d_ws, size_t ws_size,
                              hipStream_t stream) {
  (void)in_sizes; (void)n_in; (void)out_size; (void)ws_size;

  const int*   input_ids = (const int*)d_in[0];
  const float* enc_f     = (const float*)d_in[1];
  const float* hid_init  = (const float*)d_in[2];
  // d_in[3] = targets_len (unused by reference math)
  const float* emb       = (const float*)d_in[4];
  const float* W_ih0     = (const float*)d_in[5];
  const float* W_hh0     = (const float*)d_in[6];
  const float* b_ih0     = (const float*)d_in[7];
  const float* b_hh0     = (const float*)d_in[8];
  const float* W_ih1     = (const float*)d_in[9];
  const float* W_hh1     = (const float*)d_in[10];
  const float* b_ih1     = (const float*)d_in[11];
  const float* b_hh1     = (const float*)d_in[12];
  const float* W_attn    = (const float*)d_in[13];
  const float* W_l1      = (const float*)d_in[14];
  const float* W_l2      = (const float*)d_in[15];
  float* out_f = (float*)d_out;

  // ---- carve workspace (256B aligned segments) ----
  char* wp = (char*)d_ws;
  auto alloc = [&](size_t bytes) -> void* {
    void* r = (void*)wp;
    wp += (bytes + 255) & ~(size_t)255;
    return r;
  };
  __bf16* xbf     = (__bf16*)alloc((size_t)BT_ * E_ * 2);
  __bf16* Wih0b   = (__bf16*)alloc((size_t)G4_ * E_ * 2);
  __bf16* Whh0b   = (__bf16*)alloc((size_t)G4_ * H_ * 2);
  __bf16* Wih1b   = (__bf16*)alloc((size_t)G4_ * H_ * 2);
  __bf16* Whh1b   = (__bf16*)alloc((size_t)G4_ * H_ * 2);
  __bf16* Wattnb  = (__bf16*)alloc((size_t)H_ * E2_ * 2);
  __bf16* Wl1b    = (__bf16*)alloc((size_t)H_ * HC_ * 2);
  __bf16* Wl2b    = (__bf16*)alloc((size_t)VOUT_ * H_ * 2);
  __bf16* encb    = (__bf16*)alloc((size_t)B_ * S_ * E2_ * 2);
  __bf16* encTb   = (__bf16*)alloc((size_t)B_ * E2_ * S_ * 2);
  float*  Xp      = (float*)alloc((size_t)BT_ * G4_ * 4);  // reused by both layers
  __bf16* h_all0  = (__bf16*)alloc((size_t)BT_ * H_ * 2);
  __bf16* h_all1  = (__bf16*)alloc((size_t)BT_ * H_ * 2);
  __bf16* hping   = (__bf16*)alloc((size_t)B_ * H_ * 2);
  __bf16* hpong   = (__bf16*)alloc((size_t)B_ * H_ * 2);
  float*  hf32_0  = (float*)alloc((size_t)B_ * H_ * 4);
  float*  hf32_1  = (float*)alloc((size_t)B_ * H_ * 4);
  float*  cbuf    = (float*)alloc((size_t)B_ * H_ * 4);
  __bf16* aprojb  = (__bf16*)alloc((size_t)BS_ * H_ * 2);
  float*  scores  = (float*)alloc((size_t)B_ * T_ * S_ * 4);
  __bf16* attnb   = (__bf16*)alloc((size_t)B_ * T_ * S_ * 2);
  __bf16* h_c     = (__bf16*)alloc((size_t)BT_ * HC_ * 2);
  __bf16* a_l1    = (__bf16*)alloc((size_t)BT_ * H_ * 2);

  auto cast = [&](const float* s, __bf16* d, long long n) {
    cast_f32_bf16<<<(unsigned)((n + 255) / 256), 256, 0, stream>>>(s, d, n);
  };
  // generic GEMM launcher: C[b] = act(A[b] . W[b]^T); M%64==0, N%16, K%32
  auto gemm = [&](const __bf16* A, const __bf16* W, float* Cf, __bf16* Cb,
                  int M, int N, int K, int lda, int ldw, int ldcf, int ldcb,
                  long long sA, long long sW, long long sCf, long long sCb,
                  int batch, int act) {
    int ntn = N / 16;
    dim3 grid((ntn + 7) / 8, M / 64, batch);
    wmma_gemm_bf16<<<grid, 256, 0, stream>>>(A, W, Cf, Cb, M, N, K, lda, ldw,
                                             ldcf, ldcb, sA, sW, sCf, sCb, act);
  };

  // ---- 0) cast weights / encoder to bf16; gather embedding ----
  cast(W_ih0, Wih0b, (long long)G4_ * E_);
  cast(W_hh0, Whh0b, (long long)G4_ * H_);
  cast(W_ih1, Wih1b, (long long)G4_ * H_);
  cast(W_hh1, Whh1b, (long long)G4_ * H_);
  cast(W_attn, Wattnb, (long long)H_ * E2_);
  cast(W_l1, Wl1b, (long long)H_ * HC_);
  cast(W_l2, Wl2b, (long long)VOUT_ * H_);
  cast(enc_f, encb, (long long)B_ * S_ * E2_);
  embed_gather<<<BT_, E_, 0, stream>>>(input_ids, emb, xbf);

  // ---- 1) LSTM layer 0 ----
  init_state<<<(B_ * H_ + 255) / 256, 256, 0, stream>>>(hid_init, hping, hf32_0, cbuf);
  // Xp = x @ W_ih0^T  : [2048,256] x [2048,256]^T -> [2048,2048] f32 (TDM A)
  gemm(xbf, Wih0b, Xp, nullptr, BT_, G4_, E_, E_, E_, G4_, 0, 0, 0, 0, 0, 1, 0);
  {
    __bf16 *hc = hping, *hn = hpong;
    for (int t = 0; t < T_; ++t) {
      lstm_step<<<dim3(H_ / 16, B_ / 16), 32, 0, stream>>>(
          hc, hn, hf32_0, cbuf, h_all0, Xp, Whh0b, b_ih0, b_hh0, t);
      __bf16* tmp = hc; hc = hn; hn = tmp;
    }
  }

  // ---- 2) LSTM layer 1 ----
  init_state<<<(B_ * H_ + 255) / 256, 256, 0, stream>>>(hid_init + B_ * H_, hping,
                                                        hf32_1, cbuf);
  gemm(h_all0, Wih1b, Xp, nullptr, BT_, G4_, H_, H_, H_, G4_, 0, 0, 0, 0, 0, 1, 0);
  {
    __bf16 *hc = hping, *hn = hpong;
    for (int t = 0; t < T_; ++t) {
      lstm_step<<<dim3(H_ / 16, B_ / 16), 32, 0, stream>>>(
          hc, hn, hf32_1, cbuf, h_all1, Xp, Whh1b, b_ih1, b_hh1, t);
      __bf16* tmp = hc; hc = hn; hn = tmp;
    }
  }

  // ---- 3) attention ----
  // att_proj = enc @ W_attn^T : [B*S,1024] x [512,1024]^T -> bf16 [B*S,512]
  gemm(encb, Wattnb, nullptr, aprojb, BS_, H_, E2_, E2_, E2_, 0, H_,
       0, 0, 0, 0, 1, 0);
  // scores[b] = out[b] @ att_proj[b]^T : [64,512] x [64,512]^T -> [64,64] f32
  gemm(h_all1, aprojb, scores, nullptr, T_, S_, H_, H_, H_, S_, 0,
       (long long)T_ * H_, (long long)S_ * H_, (long long)T_ * S_, 0, B_, 0);
  softmax_rows64<<<B_ * T_, S_, 0, stream>>>(scores, attnb);
  transpose_enc<<<(unsigned)(((long long)B_ * S_ * E2_ + 255) / 256), 256, 0,
                  stream>>>(encb, encTb);
  // context[b] = attn[b] @ enc[b] = attn[b] . encT[b]^T -> bf16 into h_c[:,512:]
  gemm(attnb, encTb, nullptr, h_c + H_, T_, E2_, S_, S_, S_, 0, HC_,
       (long long)T_ * S_, (long long)E2_ * S_, 0, (long long)T_ * HC_, B_, 0);
  copy_out_to_hc<<<(unsigned)(((long long)BT_ * H_ + 255) / 256), 256, 0,
                   stream>>>(h_all1, h_c);

  // ---- 4) output head ----
  // a_l1 = tanh(h_c @ W_l1^T) : [2048,1536] x [512,1536]^T -> bf16 [2048,512]
  gemm(h_c, Wl1b, nullptr, a_l1, BT_, H_, HC_, HC_, HC_, 0, H_,
       0, 0, 0, 0, 1, /*tanh*/ 1);
  // logits = a_l1 @ W_l2^T straight into d_out : [2048,512] x [32000,512]^T
  gemm(a_l1, Wl2b, out_f, nullptr, BT_, VOUT_, H_, H_, H_, VOUT_, 0,
       0, 0, 0, 0, 1, 0);
  log_softmax_rows<<<BT_, 256, 0, stream>>>(out_f, VOUT_);

  // ---- 5) final hidden states [2,B,H] after the logits block ----
  write_hout<<<(B_ * H_ + 255) / 256, 256, 0, stream>>>(
      hf32_0, hf32_1, out_f + (long long)BT_ * VOUT_);
}
